// GraphNet_82420422410788
// MI455X (gfx1250) — compile-verified, matrix operand order checked by reading, not run on hardware
//
#include <hip/hip_runtime.h>
#include <hip/hip_bf16.h>

#define N_NODES 8192
#define D_IN    512
#define D_OUT   128
#define ALPHA_F 0.2f
#define NEG_INF_V -9.0e15f

#define TILE_J    32
#define ROW_BYTES 80                      // 64B of bf16 data + 16B TDM pad
#define BUF_BYTES (D_OUT * ROW_BYTES)     // 10240 B per buffer
#define NITER     (N_NODES / TILE_J)      // 256

typedef __attribute__((ext_vector_type(16))) __bf16   bf16x16;
typedef __attribute__((ext_vector_type(8)))  __bf16   bf16x8;
typedef __attribute__((ext_vector_type(8)))  float    f32x8;
typedef __attribute__((ext_vector_type(4)))  float    f32x4;
typedef __attribute__((ext_vector_type(4)))  int      i32x4;
typedef __attribute__((ext_vector_type(8)))  int      i32x8;
typedef __attribute__((ext_vector_type(4)))  unsigned u32x4;

#if __has_builtin(__builtin_amdgcn_tensor_load_to_lds)
#define USE_TDM 1
#if __has_include(<hip/amd_detail/amd_gfx1250_TDM.h>)
#warning "CDNA5-PATH: TDM enabled, 6-arg tensor_load_to_lds (therock headers)"
#else
#warning "CDNA5-PATH: TDM enabled, 5-arg tensor_load_to_lds (ROCm 7.2 form)"
#endif
#else
#warning "CDNA5-PATH: TDM builtin NOT found -- cooperative-copy fallback in use"
#endif

#ifdef USE_TDM
// ---------------------------------------------------------------------------
// TDM: DMA a 32(x, contiguous) x 128(y) bf16 tile of WhT[128][8192] into LDS.
// D# per CDNA5 ISA: data_size=2B; pad_enable with
// pad_interval=16 DWORDs (one 64B row) and pad_amount=4 DWORDs (16B) -> 80B
// LDS row stride. global_addr points at the tile start (column jb, row 0).
// ---------------------------------------------------------------------------
__device__ __forceinline__ void tdm_load_tile(unsigned lds_off, const __bf16* gptr)
{
    const unsigned long long ga = (unsigned long long)(uintptr_t)gptr;
    u32x4 g0;
    g0[0] = 1u;                                          // count=1 (valid), no gather
    g0[1] = lds_off;                                     // lds_addr (bytes)
    g0[2] = (unsigned)ga;                                // global_addr[31:0]
    g0[3] = (unsigned)((ga >> 32) & 0x01FFFFFFu) | (2u << 30); // addr[56:32] | type=2
    i32x8 g1;
    g1[0] = (1 << 16)        // data_size = 2B
          | (1 << 20)        // pad_enable
          | (3 << 22)        // pad_interval: 16 DWORDs (64B) between pads
          | (3 << 25);       // pad_amount: 4 DWORDs (16B)
    g1[1] = (int)(8192u << 16);   // tensor_dim0[15:0] in bits 63:48
    g1[2] = (int)(128u  << 16);   // tensor_dim1[15:0] in bits 95:80
    g1[3] = (int)(32u   << 16);   // tile_dim0 = 32 in bits 127:112
    g1[4] = 128;                  // tile_dim1 = 128 (bits 143:128)
    g1[5] = 8192;                 // tensor_dim0_stride[31:0]
    g1[6] = (int)(8192u << 16);   // tensor_dim1_stride[15:0] in bits 223:208
    g1[7] = 0;
    i32x4 gz = {0, 0, 0, 0};
#if __has_include(<hip/amd_detail/amd_gfx1250_TDM.h>)
    i32x8 gz8 = {0, 0, 0, 0, 0, 0, 0, 0};
    __builtin_amdgcn_tensor_load_to_lds(g0, g1, gz, gz, gz8, 0); // clang-23 6-arg form
#else
    __builtin_amdgcn_tensor_load_to_lds(g0, g1, gz, gz, 0);      // ROCm 7.2 5-arg form
#endif
}
#endif // USE_TDM

// ---------------------------------------------------------------------------
// Kernel 1: Wh = h @ W via bf16 WMMA (f32 accumulate). One wave per 16x16 tile.
// Also emits bf16 transposed copy WhT[128][8192] for the aggregation kernel.
// ---------------------------------------------------------------------------
__global__ __launch_bounds__(32)
void gat_wh_wmma(const float* __restrict__ h, const float* __restrict__ w,
                 float* __restrict__ Wh, __bf16* __restrict__ WhT)
{
    const int lane = threadIdx.x & 31;
    const int half = lane >> 4;
    const int mrow = lane & 15;
    const int r0 = blockIdx.x * 16;
    const int c0 = blockIdx.y * 16;

    const float* hrow = h + (size_t)(r0 + mrow) * D_IN;
    f32x8 acc = {0.f, 0.f, 0.f, 0.f, 0.f, 0.f, 0.f, 0.f};

    for (int kb = 0; kb < D_IN; kb += 32) {
        const int ka0 = kb + half * 8;
        const int ka1 = kb + 16 + half * 8;

        f32x4 a0 = *(const f32x4*)(hrow + ka0);
        f32x4 a1 = *(const f32x4*)(hrow + ka0 + 4);
        f32x4 a2 = *(const f32x4*)(hrow + ka1);
        f32x4 a3 = *(const f32x4*)(hrow + ka1 + 4);
        bf16x16 A;
        #pragma unroll
        for (int t = 0; t < 4; ++t) {
            A[t]      = (__bf16)a0[t];
            A[4 + t]  = (__bf16)a1[t];
            A[8 + t]  = (__bf16)a2[t];
            A[12 + t] = (__bf16)a3[t];
        }

        bf16x16 B;
        #pragma unroll
        for (int t = 0; t < 8; ++t) {
            B[t]     = (__bf16)w[(size_t)(ka0 + t) * D_OUT + c0 + mrow];
            B[8 + t] = (__bf16)w[(size_t)(ka1 + t) * D_OUT + c0 + mrow];
        }

        acc = __builtin_amdgcn_wmma_f32_16x16x32_bf16(
            false, A, false, B, (short)0, acc, false, false);
    }

    #pragma unroll
    for (int v = 0; v < 8; ++v) {
        const int row = r0 + v + 8 * half;
        const int col = c0 + mrow;
        Wh[(size_t)row * D_OUT + col] = acc[v];
        WhT[(size_t)col * N_NODES + row] = (__bf16)acc[v];
    }
}

// ---------------------------------------------------------------------------
// Kernel 2: per-row attention scalars  s[i] = Wh[i,:]·a1,  t[i] = Wh[i,:]·a2
// ---------------------------------------------------------------------------
__global__ void gat_attn_scores(const float* __restrict__ Wh,
                                const float* __restrict__ aw,
                                float* __restrict__ sarr, float* __restrict__ tarr)
{
    const int i = blockIdx.x * blockDim.x + threadIdx.x;
    if (i >= N_NODES) return;
    const float* row = Wh + (size_t)i * D_OUT;
    float s = 0.f, t = 0.f;
    #pragma unroll 8
    for (int d = 0; d < D_OUT; ++d) {
        const float v = row[d];
        s = fmaf(v, aw[d], s);
        t = fmaf(v, aw[D_OUT + d], t);
    }
    sarr[i] = s;
    tarr[i] = t;
}

// ---------------------------------------------------------------------------
// Kernel 3: fused masked-softmax aggregation (flash-attention style).
// 4 waves per block, each owning 16 rows x full D_OUT accumulator (64 VGPRs).
// Per 32-j block the WhT tile (128x32 bf16) is TDM-DMA'd into LDS once,
// double-buffered on TENSORcnt, and shared by all 4 waves (4x less L2 traffic).
// adj (268MB, read once, > L2) is streamed with non-temporal loads so the L2
// stays reserved for the hot WhT / tarr working set.
// ---------------------------------------------------------------------------
__global__ __launch_bounds__(128)
void gat_attn_aggregate(const int* __restrict__ adj,
                        const __bf16* __restrict__ WhT,
                        const float* __restrict__ sarr,
                        const float* __restrict__ tarr,
                        float* __restrict__ out)
{
    __shared__ __align__(16) char smem[2 * BUF_BYTES];

    const int tid  = threadIdx.x;
    const int wv   = tid >> 5;
    const int lane = tid & 31;
    const int half = lane >> 4;
    const int mrow = lane & 15;
    const int r0   = blockIdx.x * 64 + wv * 16;

    const float s_i = sarr[r0 + mrow];
    const int* adjrow = adj + (size_t)(r0 + mrow) * N_NODES;

    float m = -__builtin_inff();
    float lsum = 0.f;
    f32x8 acc[8];
    #pragma unroll
    for (int t = 0; t < 8; ++t)
        acc[t] = (f32x8){0.f, 0.f, 0.f, 0.f, 0.f, 0.f, 0.f, 0.f};

#ifdef USE_TDM
    const unsigned lds_base = (unsigned)(uintptr_t)(&smem[0]);
    if (wv == 0) tdm_load_tile(lds_base, WhT);   // prologue: buffer 0 <- jb=0
#endif

    for (int it = 0; it < NITER; ++it) {
        const int jb = it * TILE_J;
        const int j0 = jb + half * 8;
        const int j1 = jb + 16 + half * 8;

#ifdef USE_TDM
        if (wv == 0) {
            if (it + 1 < NITER) {
                // issue next tile into the other buffer (freed by the trailing
                // barrier of iteration it-1), then wait for the current one
                tdm_load_tile(lds_base + ((unsigned)((it + 1) & 1)) * BUF_BYTES,
                              WhT + (size_t)(jb + TILE_J));
                __builtin_amdgcn_s_wait_tensorcnt((short)1);
            } else {
                __builtin_amdgcn_s_wait_tensorcnt((short)0);
            }
        }
        __syncthreads();                       // current buffer visible to all
        const char* sbuf = smem + (it & 1) * BUF_BYTES;
#else
        __syncthreads();                       // previous reads done
        {
            const __bf16* src = WhT + (size_t)tid * N_NODES + jb;
            #pragma unroll
            for (int c = 0; c < 4; ++c)
                *(bf16x8*)(smem + tid * ROW_BYTES + c * 16) = *(const bf16x8*)(src + c * 8);
        }
        __syncthreads();
        const char* sbuf = smem;
#endif

        // adj: one-shot 268MB stream -> non-temporal (keep L2 for WhT/tarr)
        i32x4 m0 = __builtin_nontemporal_load((const i32x4*)(adjrow + j0));
        i32x4 m1 = __builtin_nontemporal_load((const i32x4*)(adjrow + j0 + 4));
        i32x4 m2 = __builtin_nontemporal_load((const i32x4*)(adjrow + j1));
        i32x4 m3 = __builtin_nontemporal_load((const i32x4*)(adjrow + j1 + 4));
        f32x4 t0 = *(const f32x4*)(tarr + j0);
        f32x4 t1 = *(const f32x4*)(tarr + j0 + 4);
        f32x4 t2 = *(const f32x4*)(tarr + j1);
        f32x4 t3 = *(const f32x4*)(tarr + j1 + 4);

        float e[16];
        #pragma unroll
        for (int t = 0; t < 4; ++t) {
            float x;
            x = s_i + t0[t]; x = (x > 0.f) ? x : ALPHA_F * x; e[t]      = (m0[t] > 0) ? x : NEG_INF_V;
            x = s_i + t1[t]; x = (x > 0.f) ? x : ALPHA_F * x; e[4 + t]  = (m1[t] > 0) ? x : NEG_INF_V;
            x = s_i + t2[t]; x = (x > 0.f) ? x : ALPHA_F * x; e[8 + t]  = (m2[t] > 0) ? x : NEG_INF_V;
            x = s_i + t3[t]; x = (x > 0.f) ? x : ALPHA_F * x; e[12 + t] = (m3[t] > 0) ? x : NEG_INF_V;
        }

        // online max across this j-block (half-lane pair shares a row)
        float em = e[0];
        #pragma unroll
        for (int t = 1; t < 16; ++t) em = fmaxf(em, e[t]);
        em = fmaxf(em, __shfl_xor(em, 16, 32));
        const float mnew = fmaxf(m, em);
        const float scale = __expf(m - mnew);      // 1.0 when row max unchanged
        m = mnew;
        lsum *= scale;

        if (__ballot(scale < 1.f)) {               // wave-uniform: EXEC stays full
            #pragma unroll
            for (int v = 0; v < 8; ++v) {
                const float sc = __shfl(scale, v + 8 * half, 32);
                #pragma unroll
                for (int tl = 0; tl < 8; ++tl) acc[tl][v] *= sc;
            }
        }

        // P = exp(e - m) as bf16 A-matrix (16x32)
        bf16x16 P;
        #pragma unroll
        for (int t = 0; t < 16; ++t) {
            const float p = __expf(e[t] - m);
            lsum += p;
            P[t] = (__bf16)p;
        }

        // acc[tile] += P (16x32) @ LDS WhT tile (32x16) -- 8 WMMAs over D_OUT
        #pragma unroll
        for (int tile = 0; tile < 8; ++tile) {
            const char* srow = sbuf + (tile * 16 + mrow) * ROW_BYTES;
            bf16x8 blo = *(const bf16x8*)(srow + 16 * half);
            bf16x8 bhi = *(const bf16x8*)(srow + 32 + 16 * half);
            bf16x16 B = __builtin_shufflevector(blo, bhi,
                0, 1, 2, 3, 4, 5, 6, 7, 8, 9, 10, 11, 12, 13, 14, 15);
            acc[tile] = __builtin_amdgcn_wmma_f32_16x16x32_bf16(
                false, P, false, B, (short)0, acc[tile], false, false);
        }

#ifdef USE_TDM
        __syncthreads();   // all waves done with this buffer before it is reused
#endif
    }

    // finalize: divide by per-row exp-sum and store (streamed, non-temporal)
    const float ltot = lsum + __shfl_xor(lsum, 16, 32);
    const float linv = 1.f / ltot;
    #pragma unroll
    for (int v = 0; v < 8; ++v) {
        const float li = __shfl(linv, v + 8 * half, 32);
        const int row = r0 + v + 8 * half;
        #pragma unroll
        for (int tile = 0; tile < 8; ++tile) {
            __builtin_nontemporal_store(acc[tile][v] * li,
                                        out + (size_t)row * D_OUT + tile * 16 + mrow);
        }
    }
}

// ---------------------------------------------------------------------------
extern "C" void kernel_launch(void* const* d_in, const int* in_sizes, int n_in,
                              void* d_out, int out_size, void* d_ws, size_t ws_size,
                              hipStream_t stream)
{
    const float* h   = (const float*)d_in[0];
    const int*   adj = (const int*)d_in[1];
    const float* w   = (const float*)d_in[2];
    const float* aw  = (const float*)d_in[3];
    float* out = (float*)d_out;

    // workspace: Wh f32 (4MB) | WhT bf16 transposed (2MB) | s | t
    float*  Wh   = (float*)d_ws;
    __bf16* WhT  = (__bf16*)(Wh + (size_t)N_NODES * D_OUT);
    float*  sarr = (float*)(WhT + (size_t)D_OUT * N_NODES);
    float*  tarr = sarr + N_NODES;

    gat_wh_wmma<<<dim3(N_NODES / 16, D_OUT / 16), 32, 0, stream>>>(h, w, Wh, WhT);
    gat_attn_scores<<<dim3(N_NODES / 256), 256, 0, stream>>>(Wh, aw, sarr, tarr);
    gat_attn_aggregate<<<dim3(N_NODES / 64), 128, 0, stream>>>(adj, WhT, sarr, tarr, out);
}
